// MultiHeadAttention_80977313398935
// MI455X (gfx1250) — compile-verified, matrix operand order checked by reading, not run on hardware
//
#include <hip/hip_runtime.h>
#include <hip/hip_bf16.h>

typedef __attribute__((ext_vector_type(16))) _Float16 v16h;
typedef __attribute__((ext_vector_type(8)))  _Float16 v8h;
typedef __attribute__((ext_vector_type(8)))  float    v8f;

#define S_LEN   4096
#define D_MODEL 512
#define NHEAD   8
#define HDIM    64

// Load one 16-half WMMA fragment: two contiguous 16B chunks.
// Caller passes p = row_base + k0 + (lane>>4)*8.
// Element j<8  -> K = k0 + (lane>>4)*8 + j
// Element j>=8 -> K = k0 + 16 + (lane>>4)*8 + (j-8)
__device__ __forceinline__ v16h load_frag16(const _Float16* p) {
  v8h a = *(const v8h*)p;
  v8h b = *(const v8h*)(p + 16);
  v16h r;
#pragma unroll
  for (int i = 0; i < 8; ++i) { r[i] = a[i]; r[i + 8] = b[i]; }
  return r;
}

__device__ __forceinline__ v8f wmma_f32_f16(v16h a, v16h b, v8f c) {
  return __builtin_amdgcn_wmma_f32_16x16x32_f16(false, a, false, b, (short)0, c,
                                                false, false);
}

// ---------------- format conversion kernels ----------------
__global__ void cvt_f32_f16(const float* __restrict__ src,
                            _Float16* __restrict__ dst, int n) {
  for (int i = blockIdx.x * blockDim.x + threadIdx.x; i < n;
       i += gridDim.x * blockDim.x)
    dst[i] = (_Float16)src[i];
}

// W[Kd][Nd] (row-major f32) -> Wt[Nd][Kd] f16
__global__ void transpose_cvt(const float* __restrict__ W,
                              _Float16* __restrict__ Wt, int Kd, int Nd) {
  int i = blockIdx.x * blockDim.x + threadIdx.x;
  if (i < Kd * Nd) {
    int k = i / Nd, n = i - k * Nd;
    Wt[(size_t)n * Kd + k] = (_Float16)W[i];
  }
}

// ---------------- WMMA GEMM: C = A[MxK] * Bt[NxK]^T + bias ----------------
// Each wave computes a 32(M) x 64(N) tile: 8 WMMA per k-step, B fragments each
// feed two WMMAs. Fragments are double-buffered: next k-step's 6 loads are in
// flight while the current 8 WMMAs execute (grid is ~1 wave/SIMD, so in-wave
// overlap, not occupancy, hides the load latency).
__global__ void gemm32x64(const _Float16* __restrict__ A,
                          const _Float16* __restrict__ Bt,
                          const float* __restrict__ bias,
                          _Float16* __restrict__ Ch, float* __restrict__ Cf,
                          int M, int N, int K, int transposed) {
  const int lane = threadIdx.x & 31;
  const int wave = blockIdx.x * (blockDim.x >> 5) + (threadIdx.x >> 5);
  const int tilesN = N >> 6;
  const int m0 = (wave / tilesN) << 5;
  const int n0 = (wave % tilesN) << 6;
  if (m0 >= M) return;
  const int lo = lane & 15, hi = lane >> 4;

  v8f acc[2][4] = {};
  const _Float16* a0p = A + (size_t)(m0 + lo) * K + hi * 8;
  const _Float16* a1p = a0p + (size_t)16 * K;
  const _Float16* bp  = Bt + (size_t)(n0 + lo) * K + hi * 8;
  const size_t bs = (size_t)16 * K;

  v16h a0 = load_frag16(a0p);
  v16h a1 = load_frag16(a1p);
  v16h b0 = load_frag16(bp);
  v16h b1 = load_frag16(bp + bs);
  v16h b2 = load_frag16(bp + 2 * bs);
  v16h b3 = load_frag16(bp + 3 * bs);

  for (int k0 = 32; k0 < K; k0 += 32) {
    __builtin_prefetch(a0p + k0 + 32, 0, 3);  // near-scope prefetch, 1 step ahead
    __builtin_prefetch(a1p + k0 + 32, 0, 3);
    __builtin_prefetch(bp + k0 + 32, 0, 3);
    v16h a0n = load_frag16(a0p + k0);
    v16h a1n = load_frag16(a1p + k0);
    v16h b0n = load_frag16(bp + k0);
    v16h b1n = load_frag16(bp + bs + k0);
    v16h b2n = load_frag16(bp + 2 * bs + k0);
    v16h b3n = load_frag16(bp + 3 * bs + k0);
    acc[0][0] = wmma_f32_f16(a0, b0, acc[0][0]);
    acc[1][0] = wmma_f32_f16(a1, b0, acc[1][0]);
    acc[0][1] = wmma_f32_f16(a0, b1, acc[0][1]);
    acc[1][1] = wmma_f32_f16(a1, b1, acc[1][1]);
    acc[0][2] = wmma_f32_f16(a0, b2, acc[0][2]);
    acc[1][2] = wmma_f32_f16(a1, b2, acc[1][2]);
    acc[0][3] = wmma_f32_f16(a0, b3, acc[0][3]);
    acc[1][3] = wmma_f32_f16(a1, b3, acc[1][3]);
    a0 = a0n; a1 = a1n; b0 = b0n; b1 = b1n; b2 = b2n; b3 = b3n;
  }
  // last k-step
  acc[0][0] = wmma_f32_f16(a0, b0, acc[0][0]);
  acc[1][0] = wmma_f32_f16(a1, b0, acc[1][0]);
  acc[0][1] = wmma_f32_f16(a0, b1, acc[0][1]);
  acc[1][1] = wmma_f32_f16(a1, b1, acc[1][1]);
  acc[0][2] = wmma_f32_f16(a0, b2, acc[0][2]);
  acc[1][2] = wmma_f32_f16(a1, b2, acc[1][2]);
  acc[0][3] = wmma_f32_f16(a0, b3, acc[0][3]);
  acc[1][3] = wmma_f32_f16(a1, b3, acc[1][3]);

#pragma unroll
  for (int half = 0; half < 2; ++half) {
    const int mb = m0 + half * 16;
#pragma unroll
    for (int t = 0; t < 4; ++t) {
      int n = n0 + t * 16 + lo;
      float bv = bias[n];
      if (Cf) {
#pragma unroll
        for (int r = 0; r < 8; ++r)
          Cf[(size_t)(mb + 8 * hi + r) * N + n] = acc[half][t][r] + bv;
      } else if (!transposed) {
#pragma unroll
        for (int r = 0; r < 8; ++r)
          Ch[(size_t)(mb + 8 * hi + r) * N + n] = (_Float16)(acc[half][t][r] + bv);
      } else {  // C[n][m]: C-frag rows are contiguous in m -> one b128 store
        v8h tmp;
#pragma unroll
        for (int r = 0; r < 8; ++r) tmp[r] = (_Float16)(acc[half][t][r] + bv);
        *(v8h*)(Ch + (size_t)n * M + mb + 8 * hi) = tmp;
      }
    }
  }
}

// ---------------- flash attention (causal), one wave = one head x 16 queries
// Qh,Kh: [S][D_MODEL] f16 row-major; Vt: [D_MODEL][S] f16; Oh: [S][D_MODEL] f16
// K fragments are double-buffered across kv blocks; V fragments and the next
// K block's loads are issued before the softmax VALU section so the exp/max
// work covers their latency.
__global__ void flash_attn(const _Float16* __restrict__ Qh,
                           const _Float16* __restrict__ Kh,
                           const _Float16* __restrict__ Vt,
                           _Float16* __restrict__ Oh) {
  const int lane = threadIdx.x & 31;
  const int wave = blockIdx.x * (blockDim.x >> 5) + (threadIdx.x >> 5);
  const int qtiles = S_LEN / 16;
  const int h = wave / qtiles;
  const int q0 = (wave % qtiles) * 16;
  const int lo = lane & 15, hi = lane >> 4;
  const int q = q0 + lo;  // softmax row owned by this lane (replicated at lo+16)

  // Q fragments (B operand of S^T = K * Q^T), loop invariant
  const _Float16* qrow = Qh + (size_t)(q0 + lo) * D_MODEL + h * HDIM + hi * 8;
  v16h qb0 = load_frag16(qrow);        // head dims 0..31
  v16h qb1 = load_frag16(qrow + 32);   // head dims 32..63

  v8f o0 = {}, o1 = {}, o2 = {}, o3 = {};
  float mrow = -1e30f, lrow = 0.0f;

  const _Float16* krow = Kh + (size_t)lo * D_MODEL + h * HDIM + hi * 8;
  const _Float16* vrow = Vt + (size_t)(h * HDIM + lo) * S_LEN + hi * 8;

  // preload K fragments for kv block 0
  v16h k00 = load_frag16(krow);
  v16h k01 = load_frag16(krow + 32);
  v16h k10 = load_frag16(krow + (size_t)16 * D_MODEL);
  v16h k11 = load_frag16(krow + (size_t)16 * D_MODEL + 32);

  const int kend = q0 + 16;  // causal: only kv blocks intersecting k <= q0+15
  for (int kb = 0; kb < kend; kb += 32) {
    // S^T tiles: s0 covers k = kb..kb+15, s1 covers k = kb+16..kb+31
    v8f s0 = {}, s1 = {};
    s0 = wmma_f32_f16(k00, qb0, s0);
    s0 = wmma_f32_f16(k01, qb1, s0);
    s1 = wmma_f32_f16(k10, qb0, s1);
    s1 = wmma_f32_f16(k11, qb1, s1);

    // issue this block's V loads and the next block's K loads now; the
    // softmax VALU work below hides their latency
    const _Float16* vb = vrow + kb;
    v16h vf0 = load_frag16(vb);
    v16h vf1 = load_frag16(vb + (size_t)16 * S_LEN);
    v16h vf2 = load_frag16(vb + (size_t)32 * S_LEN);
    v16h vf3 = load_frag16(vb + (size_t)48 * S_LEN);
    if (kb + 32 < kend) {
      const _Float16* krn = krow + (size_t)(kb + 32) * D_MODEL;
      __builtin_prefetch(krn + (size_t)32 * D_MODEL, 0, 3);
      __builtin_prefetch(vb + 64, 0, 3);
      k00 = load_frag16(krn);
      k01 = load_frag16(krn + 32);
      k10 = load_frag16(krn + (size_t)16 * D_MODEL);
      k11 = load_frag16(krn + (size_t)16 * D_MODEL + 32);
    }

    // scale + causal mask; per-lane max over its 16 entries
    float sc[16];
    float pm = -1e30f;
#pragma unroll
    for (int r = 0; r < 8; ++r) {
      int k = kb + 8 * hi + r;
      float v = s0[r] * 0.125f + ((k > q) ? -1e9f : 0.0f);
      sc[r] = v; pm = fmaxf(pm, v);
      k = kb + 16 + 8 * hi + r;
      v = s1[r] * 0.125f + ((k > q) ? -1e9f : 0.0f);
      sc[r + 8] = v; pm = fmaxf(pm, v);
    }
    // combine lane pair (l, l+16): together they hold all 32 k for row q
    pm = fmaxf(pm, __shfl_xor(pm, 16, 32));
    float mnew = fmaxf(mrow, pm);
    float alpha = __expf(mrow - mnew);
    mrow = mnew;

    // P = exp(S - m), already in the A-fragment layout for the PV WMMA
    v16h pf;
    float ps = 0.0f;
#pragma unroll
    for (int r = 0; r < 16; ++r) {
      float p = __expf(sc[r] - mnew);
      ps += p;
      pf[r] = (_Float16)p;
    }
    ps += __shfl_xor(ps, 16, 32);
    lrow = lrow * alpha + ps;

    // rescale O accumulators: accum row r belongs to query 8*hi+r
#pragma unroll
    for (int r = 0; r < 8; ++r) {
      float ar = __shfl(alpha, 8 * hi + r, 32);
      o0[r] *= ar; o1[r] *= ar; o2[r] *= ar; o3[r] *= ar;
    }

    // O += P * V
    o0 = wmma_f32_f16(pf, vf0, o0);
    o1 = wmma_f32_f16(pf, vf1, o1);
    o2 = wmma_f32_f16(pf, vf2, o2);
    o3 = wmma_f32_f16(pf, vf3, o3);
  }

  // finalize: divide by row sum, store f16 [S][D_MODEL]
#pragma unroll
  for (int r = 0; r < 8; ++r) {
    float inv = 1.0f / __shfl(lrow, 8 * hi + r, 32);
    size_t row = (size_t)(q0 + 8 * hi + r) * D_MODEL + h * HDIM + lo;
    Oh[row + 0]  = (_Float16)(o0[r] * inv);
    Oh[row + 16] = (_Float16)(o1[r] * inv);
    Oh[row + 32] = (_Float16)(o2[r] * inv);
    Oh[row + 48] = (_Float16)(o3[r] * inv);
  }
}

extern "C" void kernel_launch(void* const* d_in, const int* in_sizes, int n_in,
                              void* d_out, int out_size, void* d_ws, size_t ws_size,
                              hipStream_t stream) {
  (void)in_sizes; (void)n_in; (void)out_size; (void)ws_size;
  const float* query = (const float*)d_in[0];
  const float* value = (const float*)d_in[1];
  // d_in[2] is the causal mask: reproduced analytically, never read (saves 67MB)
  const float* wq_k  = (const float*)d_in[3];
  const float* wq_b  = (const float*)d_in[4];
  const float* wkv_k = (const float*)d_in[5];
  const float* wkv_b = (const float*)d_in[6];
  const float* wo_k  = (const float*)d_in[7];
  const float* wo_b  = (const float*)d_in[8];
  float* out = (float*)d_out;

  // workspace carve-up (all chunks 256B-aligned by construction)
  _Float16* qx   = (_Float16*)d_ws;                 // f16(query)   [S][D]
  _Float16* vx   = qx + (size_t)S_LEN * D_MODEL;    // f16(value)   [S][D]
  _Float16* wqT  = vx + (size_t)S_LEN * D_MODEL;    // wq^T         [D][D]
  _Float16* wkvT = wqT + (size_t)D_MODEL * D_MODEL; // wkv^T        [2D][D]
  _Float16* woT  = wkvT + (size_t)2 * D_MODEL * D_MODEL; // wo^T    [D][D]
  _Float16* Qh   = woT + (size_t)D_MODEL * D_MODEL; // Q proj       [S][D]
  _Float16* Kh   = Qh + (size_t)S_LEN * D_MODEL;    // K proj       [S][D]
  _Float16* Vt   = Kh + (size_t)S_LEN * D_MODEL;    // V proj^T     [D][S]
  _Float16* Oh   = Vt + (size_t)S_LEN * D_MODEL;    // attn out     [S][D]

  const int nx = S_LEN * D_MODEL;
  cvt_f32_f16<<<1024, 256, 0, stream>>>(query, qx, nx);
  cvt_f32_f16<<<1024, 256, 0, stream>>>(value, vx, nx);
  transpose_cvt<<<(D_MODEL * D_MODEL + 255) / 256, 256, 0, stream>>>(
      wq_k, wqT, D_MODEL, D_MODEL);
  transpose_cvt<<<(D_MODEL * 2 * D_MODEL + 255) / 256, 256, 0, stream>>>(
      wkv_k, wkvT, D_MODEL, 2 * D_MODEL);
  transpose_cvt<<<(D_MODEL * D_MODEL + 255) / 256, 256, 0, stream>>>(
      wo_k, woT, D_MODEL, D_MODEL);

  // projections: (S/32)*(D/64) = 1024 waves, 8 waves (256 threads) per block
  const int gemmBlocks = (S_LEN / 32) * (D_MODEL / 64) / 8;  // 128
  gemm32x64<<<gemmBlocks, 256, 0, stream>>>(qx, wqT, wq_b, Qh, nullptr,
                                            S_LEN, D_MODEL, D_MODEL, 0);
  gemm32x64<<<gemmBlocks, 256, 0, stream>>>(vx, wkvT, wkv_b, Kh, nullptr,
                                            S_LEN, D_MODEL, D_MODEL, 0);
  gemm32x64<<<gemmBlocks, 256, 0, stream>>>(
      vx, wkvT + (size_t)D_MODEL * D_MODEL, wkv_b + D_MODEL, Vt, nullptr,
      S_LEN, D_MODEL, D_MODEL, 1);  // transposed store -> Vt[d][s]

  // attention: 8 heads * 256 q-tiles = 2048 waves
  flash_attn<<<NHEAD * (S_LEN / 16) / 8, 256, 0, stream>>>(Qh, Kh, Vt, Oh);

  // output projection, f32 result + bias
  gemm32x64<<<gemmBlocks, 256, 0, stream>>>(Oh, woT, wo_b, nullptr, out,
                                            S_LEN, D_MODEL, D_MODEL, 0);
}
